// SelfAttention_2886218023414
// MI455X (gfx1250) — matrix operand order repaired and checked by reading, not compiled, not run on hardware
//
#include <hip/hip_runtime.h>
#include <hip/hip_bf16.h>

#define BDIM 8
#define SDIM 2048
#define HDIM 512
#define KT   32      // keys per flash-attention tile
#define KP   520     // LDS pitch (elems) of row-major key tile  (bank stride 4 -> conflict free)
#define TP   40      // LDS pitch of transposed V half-tile [h][t] (bank stride 20 -> conflict free)
#define PP   40      // LDS pitch of per-wave P tile

typedef __attribute__((ext_vector_type(16))) __bf16 v16bf;
typedef __attribute__((ext_vector_type(8)))  float  v8f;
typedef __attribute__((ext_vector_type(4)))  unsigned int v4u;
typedef __attribute__((ext_vector_type(8)))  int    v8i;
typedef __attribute__((ext_vector_type(4)))  int    v4i;

union FragBF { v16bf v; unsigned int u[8]; };
union FragF  { v8f   v; float        f[8]; };

static __device__ inline unsigned short f32_to_bf16_rne(float f) {
  unsigned int u = __float_as_uint(f);
  unsigned int r = u + 0x7FFFu + ((u >> 16) & 1u);
  return (unsigned short)(r >> 16);
}

#if __has_builtin(__builtin_amdgcn_tensor_load_to_lds) && \
    __has_builtin(__builtin_amdgcn_s_wait_tensorcnt)
#define USE_TDM 1
typedef __attribute__((address_space(3))) char lds_char;
static __device__ inline unsigned lds_addr_of(void* p) {
  return (unsigned)(unsigned long long)(lds_char*)p;
}
#else
#define USE_TDM 0
#endif

// ---------------------------------------------------------------------------
// Kernel 1: fp32 -> bf16 conversion (vectorized x4)
// ---------------------------------------------------------------------------
__global__ void k_cvt_bf16(const float* __restrict__ src,
                           unsigned short* __restrict__ dst, int n4) {
  int i = blockIdx.x * blockDim.x + threadIdx.x;
  int stride = gridDim.x * blockDim.x;
  for (; i < n4; i += stride) {
    float4 v = ((const float4*)src)[i];
    unsigned int lo = (unsigned int)f32_to_bf16_rne(v.x) |
                      ((unsigned int)f32_to_bf16_rne(v.y) << 16);
    unsigned int hi = (unsigned int)f32_to_bf16_rne(v.z) |
                      ((unsigned int)f32_to_bf16_rne(v.w) << 16);
    ((uint2*)dst)[i] = make_uint2(lo, hi);
  }
}

// ---------------------------------------------------------------------------
// Kernel 2: Qb = bf16( tanh( O @ W^T + bias ) )
//   One wave computes a 16(M) x 32(N) tile; K-loop of 32; fragments load as
//   b128 pairs straight from row-major bf16 (verified in round-1 codegen).
// ---------------------------------------------------------------------------
__global__ void __launch_bounds__(256) k_qgemm(
    const unsigned short* __restrict__ Ob,   // (B*S, H) bf16
    const unsigned short* __restrict__ Wb,   // (H, H)  bf16
    const float* __restrict__ bias,          // (H)     fp32
    unsigned short* __restrict__ Qb)         // (B*S, H) bf16
{
  const int lane = threadIdx.x & 31;
  const int ln   = lane & 15;
  const int half = lane >> 4;
  const int wid  = threadIdx.x >> 5;
  const int gw    = blockIdx.x * 8 + wid;
  const int mTile = gw >> 4;
  const int nTile = gw & 15;
  const int c0 = nTile * 32 + ln;
  const int c1 = c0 + 16;

  FragF acc0, acc1;
#pragma unroll
  for (int r = 0; r < 8; ++r) { acc0.f[r] = 0.f; acc1.f[r] = 0.f; }

  const unsigned short* Arow = Ob + (size_t)(mTile * 16 + ln) * HDIM;
  const unsigned short* B0   = Wb + (size_t)c0 * HDIM;
  const unsigned short* B1   = Wb + (size_t)c1 * HDIM;

  for (int k0 = 0; k0 < HDIM; k0 += 32) {
    FragBF a, b0, b1;
#pragma unroll
    for (int j = 0; j < 4; ++j) {
      int k = k0 + 8 * half + 2 * j;
      a.u[j]     = *(const unsigned int*)(Arow + k);
      a.u[j + 4] = *(const unsigned int*)(Arow + k + 16);
    }
#pragma unroll
    for (int j = 0; j < 8; ++j) {
      int k = k0 + 8 * half + 2 * (j & 3) + 16 * (j >> 2);
      b0.u[j] = *(const unsigned int*)(B0 + k);
      b1.u[j] = *(const unsigned int*)(B1 + k);
    }
    acc0.v = __builtin_amdgcn_wmma_f32_16x16x32_bf16(false, a.v, false, b0.v,
                                                     (short)0, acc0.v, false, false);
    acc1.v = __builtin_amdgcn_wmma_f32_16x16x32_bf16(false, a.v, false, b1.v,
                                                     (short)0, acc1.v, false, false);
  }

  const float bi0 = bias[c0], bi1 = bias[c1];
#pragma unroll
  for (int r = 0; r < 8; ++r) {
    int orow = mTile * 16 + r + 8 * half;
    Qb[(size_t)orow * HDIM + c0] = f32_to_bf16_rne(tanhf(acc0.f[r] + bi0));
    Qb[(size_t)orow * HDIM + c1] = f32_to_bf16_rne(tanhf(acc1.f[r] + bi1));
  }
}

// ---------------------------------------------------------------------------
// Kernel 3: fused flash attention: out = softmax(Q @ O^T) @ O   per batch.
//   Block = 4 waves; wave owns 16 Q rows. Per 32-key tile:
//    - TDM (TENSOR_LOAD_TO_LDS) stages the 32x512 bf16 key tile row-major
//      into LDS with a padded pitch of 520 elems (pad 4 DW per 256-DW row),
//      overlapped with a manual transposed staging of the first V half-tile.
//    - scores via 32 WMMAs reading B fragments from padded LDS (conflict-free)
//    - online softmax with 16-lane shfl_xor reductions
//    - P relayout through per-wave LDS; Oa accumulation in two 256-col halves
// ---------------------------------------------------------------------------
__global__ void __launch_bounds__(128) k_attn(
    const unsigned short* __restrict__ Ob,   // (B, S, H) bf16
    const unsigned short* __restrict__ Qb,   // (B, S, H) bf16
    float* __restrict__ out)                 // (B, S, H) fp32
{
  __shared__ unsigned short kS[KT * KP];          // row-major key tile [t][h], padded
  __shared__ unsigned short kT[256 * TP];         // transposed V half-tile [h'][t]
  __shared__ unsigned short pS[4][16 * PP];       // per-wave P tile [row][key]

  const int tid  = threadIdx.x;
  const int lane = tid & 31;
  const int ln   = lane & 15;
  const int half = lane >> 4;
  const int wid  = tid >> 5;

  const int b  = blockIdx.x >> 5;                 // 32 blocks per batch
  const int m0 = (blockIdx.x & 31) * 64 + wid * 16;

  const unsigned short* ObB  = Ob + (size_t)b * SDIM * HDIM;
  const unsigned short* Qrow = Qb + ((size_t)b * SDIM + m0 + ln) * HDIM;

  // Cache the wave's 16x512 Q tile as 16 A-fragments
  FragBF qf[16];
#pragma unroll
  for (int k = 0; k < 16; ++k) {
    int k0 = k * 32;
#pragma unroll
    for (int j = 0; j < 4; ++j) {
      int kk = k0 + 8 * half + 2 * j;
      qf[k].u[j]     = *(const unsigned int*)(Qrow + kk);
      qf[k].u[j + 4] = *(const unsigned int*)(Qrow + kk + 16);
    }
  }

  FragF acc[32];
#pragma unroll
  for (int j = 0; j < 32; ++j)
#pragma unroll
    for (int r = 0; r < 8; ++r) acc[j].f[r] = 0.f;

  float m_run[8], l_run[8];
#pragma unroll
  for (int r = 0; r < 8; ++r) { m_run[r] = -1e30f; l_run[r] = 0.f; }

  for (int t0 = 0; t0 < SDIM; t0 += KT) {
    __syncthreads();                              // protect previous tile reads

    // ---- stage row-major key tile into kS (padded pitch KP) ----
#if USE_TDM
    if (wid == 0) {
      unsigned long long ga =
          (unsigned long long)(ObB + (size_t)t0 * HDIM);
      unsigned ldsa = __builtin_amdgcn_readfirstlane((int)lds_addr_of(kS));
      unsigned galo = __builtin_amdgcn_readfirstlane((int)(unsigned)ga);
      unsigned gahi = __builtin_amdgcn_readfirstlane((int)(unsigned)(ga >> 32));
      v4u g0;
      g0[0] = 1u;                                  // count=1, user descriptor
      g0[1] = ldsa;                                // lds_addr [63:32]
      g0[2] = galo;                                // global_addr [95:64]
      g0[3] = (gahi & 0x01FFFFFFu) | (2u << 30);   // global_addr[56:32], type=2
      v8i g1;
      g1[0] = (int)((1u << 16) |                   // data_size = 2 bytes
                    (1u << 20) |                   // pad_enable
                    (7u << 22) |                   // pad_interval: 256 DWORDs (1 row)
                    (3u << 25));                   // pad_amount: 4 DWORDs (8 elems)
      g1[1] = (int)((HDIM & 0xFFFFu) << 16);       // tensor_dim0[15:0]=512
      g1[2] = (int)((HDIM >> 16) | (KT << 16));    // tensor_dim0[31:16], tensor_dim1=32
      g1[3] = (int)((unsigned)HDIM << 16);         // tile_dim0 = 512
      g1[4] = (int)KT;                             // tile_dim1 = 32, tile_dim2 = 0
      g1[5] = (int)HDIM;                           // tensor_dim0_stride = 512
      g1[6] = 0;
      g1[7] = 0;
      v4i z4; z4[0] = z4[1] = z4[2] = z4[3] = 0;   // groups 2/3 unused (2-D)
      v8i z8;
      z8[0] = z8[1] = z8[2] = z8[3] = 0;
      z8[4] = z8[5] = z8[6] = z8[7] = 0;
      // clang-23 / amdgpu-toolchain form: 6 args (.., int32x4, int32x4, int32x8, cpol)
      __builtin_amdgcn_tensor_load_to_lds(g0, g1, z4, z4, z8, 0);
    }
#else
    for (int cc = tid; cc < KT * HDIM / 8; cc += 128) {
      int t  = cc >> 6;                            // 64 chunks of 8 per row
      int h8 = (cc & 63) << 3;
      *(uint4*)(kS + t * KP + h8) =
          *(const uint4*)(ObB + (size_t)(t0 + t) * HDIM + h8);
    }
#endif

    // ---- transposed V half-tile 0 (h = 0..255) from global, overlapping TDM ----
    for (int cc = tid; cc < 1024; cc += 128) {
      int t  = cc >> 5;                            // 32 chunks of 8 per 256-elem half
      int h8 = (cc & 31) << 3;
      uint4 w = *(const uint4*)(ObB + (size_t)(t0 + t) * HDIM + h8);
      unsigned short* d = kT + h8 * TP + t;
      d[0 * TP] = (unsigned short)w.x; d[1 * TP] = (unsigned short)(w.x >> 16);
      d[2 * TP] = (unsigned short)w.y; d[3 * TP] = (unsigned short)(w.y >> 16);
      d[4 * TP] = (unsigned short)w.z; d[5 * TP] = (unsigned short)(w.z >> 16);
      d[6 * TP] = (unsigned short)w.w; d[7 * TP] = (unsigned short)(w.w >> 16);
    }

    // prefetch next key tile toward L2
    if (t0 + KT < SDIM)
      __builtin_prefetch(ObB + (size_t)(t0 + KT + lane) * HDIM + (size_t)wid * 128, 0, 1);

#if USE_TDM
    if (wid == 0) __builtin_amdgcn_s_wait_tensorcnt(0);
#endif
    __syncthreads();

    // ---- scores: s0 = Q @ K[0..15]^T, s1 = Q @ K[16..31]^T (B frags from kS) ----
    FragF s0, s1;
#pragma unroll
    for (int r = 0; r < 8; ++r) { s0.f[r] = 0.f; s1.f[r] = 0.f; }
    const unsigned short* K0 = kS + (size_t)ln * KP;
    const unsigned short* K1 = kS + (size_t)(16 + ln) * KP;
#pragma unroll
    for (int k = 0; k < 16; ++k) {
      FragBF b0, b1;
#pragma unroll
      for (int j = 0; j < 8; ++j) {
        int kk = k * 32 + 8 * half + 2 * (j & 3) + 16 * (j >> 2);
        b0.u[j] = *(const unsigned int*)(K0 + kk);
        b1.u[j] = *(const unsigned int*)(K1 + kk);
      }
      s0.v = __builtin_amdgcn_wmma_f32_16x16x32_bf16(false, qf[k].v, false, b0.v,
                                                     (short)0, s0.v, false, false);
      s1.v = __builtin_amdgcn_wmma_f32_16x16x32_bf16(false, qf[k].v, false, b1.v,
                                                     (short)0, s1.v, false, false);
    }

    // ---- online softmax (rows r / r+8 live in lane halves) ----
    float alpha[8], p0[8], p1[8];
#pragma unroll
    for (int r = 0; r < 8; ++r) {
      float mx = fmaxf(s0.f[r], s1.f[r]);
#pragma unroll
      for (int msk = 1; msk < 16; msk <<= 1)
        mx = fmaxf(mx, __shfl_xor(mx, msk, 32));
      float mn = fmaxf(m_run[r], mx);
      alpha[r] = __expf(m_run[r] - mn);
      p0[r] = __expf(s0.f[r] - mn);
      p1[r] = __expf(s1.f[r] - mn);
      float rs = p0[r] + p1[r];
#pragma unroll
      for (int msk = 1; msk < 16; msk <<= 1)
        rs += __shfl_xor(rs, msk, 32);
      l_run[r] = l_run[r] * alpha[r] + rs;
      m_run[r] = mn;
    }
#pragma unroll
    for (int j = 0; j < 32; ++j)
#pragma unroll
      for (int r = 0; r < 8; ++r) acc[j].f[r] *= alpha[r];

    // ---- P: C-layout -> bf16 LDS -> A-layout fragment ----
    unsigned short* pw = pS[wid];
#pragma unroll
    for (int r = 0; r < 8; ++r) {
      int row = r + 8 * half;
      pw[row * PP + ln]      = f32_to_bf16_rne(p0[r]);
      pw[row * PP + 16 + ln] = f32_to_bf16_rne(p1[r]);
    }
    FragBF pa;                                     // same-wave DS: in order
#pragma unroll
    for (int j = 0; j < 4; ++j) {
      int kk = 16 * half + 2 * j;
      pa.u[j]     = *(const unsigned int*)(pw + ln * PP + kk);
      pa.u[j + 4] = *(const unsigned int*)(pw + ln * PP + kk + 8);
    }

    // ---- Oa accumulate, half 0: cols 0..255 ----
#pragma unroll
    for (int j = 0; j < 16; ++j) {
      FragBF vb;
      const unsigned short* vrow = kT + (j * 16 + ln) * TP + 16 * half;
#pragma unroll
      for (int i = 0; i < 8; ++i)
        vb.u[i] = *(const unsigned int*)(vrow + 2 * i);
      acc[j].v = __builtin_amdgcn_wmma_f32_16x16x32_bf16(false, pa.v, false, vb.v,
                                                         (short)0, acc[j].v, false, false);
    }

    // ---- transposed V half-tile 1 (h = 256..511) ----
    __syncthreads();
    for (int cc = tid; cc < 1024; cc += 128) {
      int t  = cc >> 5;
      int h8 = 256 + ((cc & 31) << 3);
      uint4 w = *(const uint4*)(ObB + (size_t)(t0 + t) * HDIM + h8);
      unsigned short* d = kT + (h8 - 256) * TP + t;
      d[0 * TP] = (unsigned short)w.x; d[1 * TP] = (unsigned short)(w.x >> 16);
      d[2 * TP] = (unsigned short)w.y; d[3 * TP] = (unsigned short)(w.y >> 16);
      d[4 * TP] = (unsigned short)w.z; d[5 * TP] = (unsigned short)(w.z >> 16);
      d[6 * TP] = (unsigned short)w.w; d[7 * TP] = (unsigned short)(w.w >> 16);
    }
    __syncthreads();

    // ---- Oa accumulate, half 1: cols 256..511 ----
#pragma unroll
    for (int j = 0; j < 16; ++j) {
      FragBF vb;
      const unsigned short* vrow = kT + (j * 16 + ln) * TP + 16 * half;
#pragma unroll
      for (int i = 0; i < 8; ++i)
        vb.u[i] = *(const unsigned int*)(vrow + 2 * i);
      acc[16 + j].v = __builtin_amdgcn_wmma_f32_16x16x32_bf16(false, pa.v, false, vb.v,
                                                              (short)0, acc[16 + j].v, false, false);
    }
  }

  // ---- epilogue: divide by softmax denominator, store fp32 ----
  float* outB = out + (size_t)b * SDIM * HDIM;
#pragma unroll
  for (int r = 0; r < 8; ++r) {
    float inv = 1.f / l_run[r];
    int row = m0 + r + 8 * half;
#pragma unroll
    for (int j = 0; j < 32; ++j)
      outB[(size_t)row * HDIM + j * 16 + ln] = acc[j].f[r] * inv;
  }
}

// ---------------------------------------------------------------------------
extern "C" void kernel_launch(void* const* d_in, const int* in_sizes, int n_in,
                              void* d_out, int out_size, void* d_ws, size_t ws_size,
                              hipStream_t stream) {
  const float* O    = (const float*)d_in[0];   // (B,S,H)
  const float* Wa_w = (const float*)d_in[1];   // (H,H)
  const float* Wa_b = (const float*)d_in[2];   // (H)
  float* out = (float*)d_out;

  const size_t nO = (size_t)BDIM * SDIM * HDIM;   // 8388608
  const size_t nW = (size_t)HDIM * HDIM;          // 262144

  unsigned short* Ob  = (unsigned short*)d_ws;    // bf16 O
  unsigned short* Qb  = Ob + nO;                  // bf16 Q
  unsigned short* Wbq = Qb + nO;                  // bf16 W
  // total ws use: (2*8388608 + 262144) * 2 bytes ~= 34 MB

  k_cvt_bf16<<<2048, 256, 0, stream>>>(O, Ob, (int)(nO / 4));
  k_cvt_bf16<<<128, 256, 0, stream>>>(Wa_w, Wbq, (int)(nW / 4));
  k_qgemm<<<2048, 256, 0, stream>>>(Ob, Wbq, Wa_b, Qb);
  k_attn<<<256, 128, 0, stream>>>(Ob, Qb, out);
}
